// FeatureDecoder_22162031247562
// MI455X (gfx1250) — matrix operand order repaired and checked by reading, not compiled, not run on hardware
//
#include <hip/hip_runtime.h>
#include <hip/hip_bf16.h>

// ---------------------------------------------------------------------------
// Types for CDNA5 WMMA (wave32): v_wmma_f32_16x16x32_bf16
// ---------------------------------------------------------------------------
typedef __bf16 v8bf  __attribute__((ext_vector_type(8)));
typedef __bf16 v16bf __attribute__((ext_vector_type(16)));
typedef float  v8f   __attribute__((ext_vector_type(8)));

__device__ __forceinline__ __bf16 f2bf(float f) {
  unsigned u = __builtin_bit_cast(unsigned, f);
  unsigned r = (u + 0x7FFFu + ((u >> 16) & 1u)) >> 16;   // round-to-nearest-even
  return __builtin_bit_cast(__bf16, (unsigned short)r);
}
__device__ __forceinline__ float bf2f(__bf16 h) {
  unsigned short s = __builtin_bit_cast(unsigned short, h);
  unsigned u = ((unsigned)s) << 16;
  return __builtin_bit_cast(float, u);
}

// ---------------------------------------------------------------------------
// CDNA5 async global->LDS copy (ASYNCcnt path), guarded with fallback.
// Builtin signature (from hipcc diagnostic): param0 = v4i* (global),
// so pass int4 pointers in AS1/AS3.
// ---------------------------------------------------------------------------
#if __has_builtin(__builtin_amdgcn_global_load_async_to_lds_b128)
#define HAVE_ASYNC_LDS 1
#else
#define HAVE_ASYNC_LDS 0
#endif

#if HAVE_ASYNC_LDS
typedef int v4i __attribute__((ext_vector_type(4)));
typedef __attribute__((address_space(1))) v4i as1_v4i;
typedef __attribute__((address_space(3))) v4i as3_v4i;
#endif

__device__ __forceinline__ void copy16_g2l(const __bf16* g, __bf16* l) {
#if HAVE_ASYNC_LDS
  __builtin_amdgcn_global_load_async_to_lds_b128((as1_v4i*)g, (as3_v4i*)l, 0, 0);
#else
  *(v8bf*)l = *(const v8bf*)g;
#endif
}
__device__ __forceinline__ void copy_wait() {
#if HAVE_ASYNC_LDS && __has_builtin(__builtin_amdgcn_s_wait_asynccnt)
  __builtin_amdgcn_s_wait_asynccnt(0);
#endif
}

// ---------------------------------------------------------------------------
// GEMM: C[M,N] = A[M,K](bf16) * Wt[N,K](bf16, pre-transposed) + bias[N]
// 256 threads = 8 wave32; block tile 128x128, K-step 32 (one WMMA K),
// double-buffered LDS with async global->LDS staging of the next K-slab
// overlapped with WMMA on the current one.
// Wave grid 4(M) x 2(N): each wave owns a 32x64 subtile = 2x4 WMMA accs.
// ---------------------------------------------------------------------------
#define BM 128
#define BN 128
#define BK 32

template<bool OUT_BF16>
__global__ __launch_bounds__(256)
void gemm_bias(const __bf16* __restrict__ A, const __bf16* __restrict__ Wt,
               const float* __restrict__ bias, void* __restrict__ Cout,
               int M, int N, int K)
{
  __shared__ __bf16 As[2][BM * BK];         // As[b][m*BK + k]
  __shared__ __bf16 Bs[2][BN * BK];         // Bs[b][n*BK + k]  (K-contig per column)

  const int tid   = threadIdx.x;
  const int lane  = tid & 31;
  const int wid   = tid >> 5;               // 0..7
  const int wm    = wid & 3;                // 4 waves along M
  const int wn    = wid >> 2;               // 2 waves along N
  const int mTile = blockIdx.y * BM;
  const int nTile = blockIdx.x * BN;
  const int hf    = lane >> 4;              // K-half selector (ISA layout)
  const int l16   = lane & 15;

  // Branchless staging: 512 x 16B chunks per tile, 2 per thread for A and B.
  auto stage = [&](int k0, int buf) {
#pragma unroll
    for (int i = 0; i < 2; i++) {
      int chunk = tid + i * 256;
      int row   = chunk >> 2;
      int kc    = (chunk & 3) << 3;
      copy16_g2l(&A [(size_t)(mTile + row) * K + k0 + kc], &As[buf][row * BK + kc]);
      copy16_g2l(&Wt[(size_t)(nTile + row) * K + k0 + kc], &Bs[buf][row * BK + kc]);
    }
  };

  v8f acc[2][4] = {};
  const int nk = K / BK;

  stage(0, 0);
  for (int t = 0; t < nk; ++t) {
    const int cur = t & 1;
    copy_wait();                 // my async copies for tile t landed in LDS
    __syncthreads();             // everyone's landed -> tile t readable
    if (t + 1 < nk) stage((t + 1) * BK, cur ^ 1);   // overlap next-slab copy

    // ---- build fragments per ISA VGPR layouts ----
    v16bf af[2];
#pragma unroll
    for (int mi = 0; mi < 2; mi++) {
      int row = wm * 32 + mi * 16 + l16;
      v8bf lo = *(const v8bf*)&As[cur][row * BK + hf * 8];        // K {0..7}/{8..15}
      v8bf hi = *(const v8bf*)&As[cur][row * BK + 16 + hf * 8];   // K {16..23}/{24..31}
#pragma unroll
      for (int e = 0; e < 8; e++) { af[mi][e] = lo[e]; af[mi][8 + e] = hi[e]; }
    }
    v16bf bfr[4];
#pragma unroll
    for (int ni = 0; ni < 4; ni++) {
      int col = wn * 64 + ni * 16 + l16;
      v8bf lo = *(const v8bf*)&Bs[cur][col * BK + hf * 8];        // K {0..7}/{8..15}
      v8bf hi = *(const v8bf*)&Bs[cur][col * BK + 16 + hf * 8];   // K {16..23}/{24..31}
#pragma unroll
      for (int e = 0; e < 8; e++) { bfr[ni][e] = lo[e]; bfr[ni][8 + e] = hi[e]; }
    }

#pragma unroll
    for (int mi = 0; mi < 2; mi++)
#pragma unroll
      for (int ni = 0; ni < 4; ni++)
        acc[mi][ni] = __builtin_amdgcn_wmma_f32_16x16x32_bf16(
            false, af[mi], false, bfr[ni], (short)0, acc[mi][ni], false, false);

    __syncthreads();             // tile t fully consumed before it is re-staged
  }

  // ---- epilogue: bias + store (fp32 or bf16) per C/D VGPR layout ----
#pragma unroll
  for (int mi = 0; mi < 2; mi++) {
    int rowBase = mTile + wm * 32 + mi * 16 + hf * 8;
#pragma unroll
    for (int ni = 0; ni < 4; ni++) {
      int col = nTile + wn * 64 + ni * 16 + l16;
      float bv = bias[col];
#pragma unroll
      for (int r = 0; r < 8; r++) {
        float v = acc[mi][ni][r] + bv;
        size_t off = (size_t)(rowBase + r) * N + col;
        if (OUT_BF16) ((__bf16*)Cout)[off] = f2bf(v);
        else          ((float*)Cout)[off]  = v;
      }
    }
  }
}

// ---------------------------------------------------------------------------
// Weight prep: transpose K x N fp32 -> N x K bf16 (done once, tiny cost)
// ---------------------------------------------------------------------------
#define TT 32
__global__ void k_transpose_f32_to_bf16(const float* __restrict__ src,
                                        __bf16* __restrict__ dst, int K, int N) {
  __shared__ float tile[TT][TT + 1];
  int kb = blockIdx.y * TT, nb = blockIdx.x * TT;
  int tx = threadIdx.x, ty = threadIdx.y;            // 32 x 8
  for (int i = ty; i < TT; i += 8)
    tile[i][tx] = src[(size_t)(kb + i) * N + nb + tx];
  __syncthreads();
  for (int i = ty; i < TT; i += 8)
    dst[(size_t)(nb + i) * K + kb + tx] = f2bf(tile[tx][i]);
}

// ---------------------------------------------------------------------------
// Elementwise / reduction helpers
// ---------------------------------------------------------------------------
#define SLABS 512

__global__ void k_f32_to_bf16(const float* __restrict__ s, __bf16* __restrict__ d, size_t n) {
  for (size_t i = blockIdx.x * (size_t)blockDim.x + threadIdx.x; i < n;
       i += (size_t)gridDim.x * blockDim.x)
    d[i] = f2bf(s[i]);
}

// partial sums over row slabs: deterministic (no float atomics)
__global__ void k_bn_stats(const __bf16* __restrict__ h, float* __restrict__ p1,
                           float* __restrict__ p2, int M, int N) {
  int c    = blockIdx.x * blockDim.x + threadIdx.x;   // channel
  int slab = blockIdx.y;
  int rows = M / SLABS;
  int r0   = slab * rows;
  float s = 0.f, s2 = 0.f;
  for (int r = r0; r < r0 + rows; ++r) {
    float v = bf2f(h[(size_t)r * N + c]);
    s += v; s2 += v * v;
  }
  p1[(size_t)slab * N + c] = s;
  p2[(size_t)slab * N + c] = s2;
}

__global__ void k_bn_finalize(const float* __restrict__ p1, const float* __restrict__ p2,
                              const float* __restrict__ g, const float* __restrict__ be,
                              float* __restrict__ scale, float* __restrict__ shift,
                              int M, int N) {
  int c = blockIdx.x * blockDim.x + threadIdx.x;
  float s = 0.f, s2 = 0.f;
  for (int i = 0; i < SLABS; i++) { s += p1[(size_t)i * N + c]; s2 += p2[(size_t)i * N + c]; }
  float mean = s / (float)M;
  float var  = s2 / (float)M - mean * mean;
  float sc   = g[c] * rsqrtf(var + 1e-5f);
  scale[c] = sc;
  shift[c] = be[c] - mean * sc;
}

__global__ void k_bn_relu(const __bf16* __restrict__ h, const float* __restrict__ scale,
                          const float* __restrict__ shift, __bf16* __restrict__ out,
                          size_t total, int Nmask) {
  for (size_t e = blockIdx.x * (size_t)blockDim.x + threadIdx.x; e < total;
       e += (size_t)gridDim.x * blockDim.x) {
    int c = (int)(e & (size_t)Nmask);
    float v = bf2f(h[e]) * scale[c] + shift[c];
    out[e] = f2bf(v > 0.f ? v : 0.f);
  }
}

// out[b,n, 0:C] = f[b,n,:] ; out[b,n, C:2C] = x[b, idx[b,n], :]   (all -> bf16)
__global__ void k_gather_concat(const float* __restrict__ f, const float* __restrict__ x,
                                const int* __restrict__ idx, __bf16* __restrict__ out,
                                int Bn, int Nf, int Nx, int C) {
  size_t total = (size_t)Bn * Nf * 2 * C;
  int C2 = 2 * C;
  for (size_t e = blockIdx.x * (size_t)blockDim.x + threadIdx.x; e < total;
       e += (size_t)gridDim.x * blockDim.x) {
    int c     = (int)(e % (size_t)C2);
    size_t rn = e / (size_t)C2;                // = b*Nf + n
    int b     = (int)(rn / (size_t)Nf);
    float v;
    if (c < C) v = f[rn * C + c];
    else       v = x[((size_t)b * Nx + idx[rn]) * C + (c - C)];
    out[e] = f2bf(v);
  }
}

// ---------------------------------------------------------------------------
// Host launcher
// ---------------------------------------------------------------------------
extern "C" void kernel_launch(void* const* d_in, const int* in_sizes, int n_in,
                              void* d_out, int out_size, void* d_ws, size_t ws_size,
                              hipStream_t stream) {
  const float* f0 = (const float*)d_in[0];
  const float* f1 = (const float*)d_in[1];
  const float* f2 = (const float*)d_in[2];
  const float* up0_w1 = (const float*)d_in[3];
  const float* up0_b1 = (const float*)d_in[4];
  const float* up0_g  = (const float*)d_in[5];
  const float* up0_be = (const float*)d_in[6];
  const float* up0_w2 = (const float*)d_in[7];
  const float* up0_b2 = (const float*)d_in[8];
  const float* up1_w1 = (const float*)d_in[9];
  const float* up1_b1 = (const float*)d_in[10];
  const float* up1_g  = (const float*)d_in[11];
  const float* up1_be = (const float*)d_in[12];
  const float* up1_w2 = (const float*)d_in[13];
  const float* up1_b2 = (const float*)d_in[14];
  const float* sk0_w1 = (const float*)d_in[15];
  const float* sk0_b1 = (const float*)d_in[16];
  const float* sk0_g  = (const float*)d_in[17];
  const float* sk0_be = (const float*)d_in[18];
  const float* sk0_w2 = (const float*)d_in[19];
  const float* sk0_b2 = (const float*)d_in[20];
  const float* sk1_w1 = (const float*)d_in[21];
  const float* sk1_b1 = (const float*)d_in[22];
  const float* sk1_g  = (const float*)d_in[23];
  const float* sk1_be = (const float*)d_in[24];
  const float* sk1_w2 = (const float*)d_in[25];
  const float* sk1_b2 = (const float*)d_in[26];
  const int*   pool0  = (const int*)d_in[27];
  const int*   pool1  = (const int*)d_in[28];

  char* ws = (char*)d_ws;
  size_t off = 0;
  auto alloc = [&](size_t bytes) -> char* {
    off = (off + 255) & ~(size_t)255;
    char* p = ws + off; off += bytes; return p;
  };

  __bf16* wUp1W1 = (__bf16*)alloc((size_t)1024 * 1024 * 2);   // transposed: N x K
  __bf16* wUp1W2 = (__bf16*)alloc((size_t)512  * 1024 * 2);
  __bf16* wSk1W1 = (__bf16*)alloc((size_t)1024 * 1024 * 2);
  __bf16* wSk1W2 = (__bf16*)alloc((size_t)512  * 1024 * 2);
  __bf16* wUp0W1 = (__bf16*)alloc((size_t)512  * 512  * 2);
  __bf16* wUp0W2 = (__bf16*)alloc((size_t)256  * 512  * 2);
  __bf16* wSk0W1 = (__bf16*)alloc((size_t)512  * 512  * 2);
  __bf16* wSk0W2 = (__bf16*)alloc((size_t)256  * 512  * 2);
  __bf16* bufA   = (__bf16*)alloc((size_t)131072 * 512 * 2);  // activations (bf16)
  __bf16* bufH   = (__bf16*)alloc((size_t)131072 * 512 * 2);  // hidden h (bf16)
  float*  bufX   = (float*) alloc((size_t)32768  * 512 * 4);  // fp32 x intermediates
  float*  part1  = (float*) alloc((size_t)SLABS * 1024 * 4);
  float*  part2  = (float*) alloc((size_t)SLABS * 1024 * 4);
  float*  scaleB = (float*) alloc((size_t)1024 * 4);
  float*  shiftB = (float*) alloc((size_t)1024 * 4);

  auto cvt = [&](const float* s, __bf16* d, size_t n) {
    k_f32_to_bf16<<<4096, 256, 0, stream>>>(s, d, n);
  };
  // weight K x N fp32  ->  N x K bf16 (so GEMM B staging is pure b128 copies)
  auto cvtT = [&](const float* s, __bf16* d, int K_, int N_) {
    k_transpose_f32_to_bf16<<<dim3(N_ / TT, K_ / TT), dim3(TT, 8), 0, stream>>>(s, d, K_, N_);
  };

  // MLP: Linear(K->Nh)+bias -> BN(batch stats)+ReLU -> Linear(Nh->No)+bias
  auto mlp = [&](const __bf16* Ain, int M, int K, int Nh, int No,
                 const __bf16* W1t, const float* b1, const float* g, const float* be,
                 const __bf16* W2t, const float* b2, __bf16* nbuf, float* outf) {
    gemm_bias<true ><<<dim3(Nh / BN, M / BM), 256, 0, stream>>>(Ain, W1t, b1, bufH, M, Nh, K);
    k_bn_stats   <<<dim3(Nh / 256, SLABS), 256, 0, stream>>>(bufH, part1, part2, M, Nh);
    k_bn_finalize<<<dim3(Nh / 256), 256, 0, stream>>>(part1, part2, g, be, scaleB, shiftB, M, Nh);
    k_bn_relu    <<<4096, 256, 0, stream>>>(bufH, scaleB, shiftB, nbuf, (size_t)M * Nh, Nh - 1);
    gemm_bias<false><<<dim3(No / BN, M / BM), 256, 0, stream>>>(nbuf, W2t, b2, outf, M, No, Nh);
  };

  // --- prep weights: transpose + convert to bf16 (L2-resident afterwards) ---
  cvtT(up1_w1, wUp1W1, 1024, 1024);
  cvtT(up1_w2, wUp1W2, 1024, 512);
  cvtT(sk1_w1, wSk1W1, 1024, 1024);
  cvtT(sk1_w2, wSk1W2, 1024, 512);
  cvtT(up0_w1, wUp0W1, 512, 512);
  cvtT(up0_w2, wUp0W2, 512, 256);
  cvtT(sk0_w1, wSk0W1, 512, 512);
  cvtT(sk0_w2, wSk0W2, 512, 256);

  // --- level 1: up1 on f2, gather via pool1, skip1 with f1 ---
  cvt(f2, bufA, (size_t)4 * 2048 * 1024);
  mlp(bufA, 4 * 2048, 1024, 1024, 512,
      wUp1W1, up1_b1, up1_g, up1_be, wUp1W2, up1_b2, bufA, bufX);     // -> (4,2048,512) fp32
  k_gather_concat<<<4096, 256, 0, stream>>>(f1, bufX, pool1, bufA, 4, 8192, 2048, 512);
  mlp(bufA, 4 * 8192, 1024, 1024, 512,
      wSk1W1, sk1_b1, sk1_g, sk1_be, wSk1W2, sk1_b2, bufA, bufX);     // -> (4,8192,512) fp32

  // --- level 0: up0, gather via pool0, skip0 with f0 -> d_out ---
  cvt(bufX, bufA, (size_t)4 * 8192 * 512);
  mlp(bufA, 4 * 8192, 512, 512, 256,
      wUp0W1, up0_b1, up0_g, up0_be, wUp0W2, up0_b2, bufA, bufX);     // -> (4,8192,256) fp32
  k_gather_concat<<<4096, 256, 0, stream>>>(f0, bufX, pool0, bufA, 4, 32768, 8192, 256);
  mlp(bufA, 4 * 32768, 512, 512, 256,
      wSk0W1, sk0_b1, sk0_g, sk0_be, wSk0W2, sk0_b2, bufA, (float*)d_out);
}
